// BldgsRmJointModel_7533372637731
// MI455X (gfx1250) — matrix operand order repaired and checked by reading, not compiled, not run on hardware
//
#include <hip/hip_runtime.h>
#include <hip/hip_bf16.h>
#include <math.h>

// ---------------------------------------------------------------------------
// GraphSAGE (max-aggr) pipeline for MI455X / gfx1250.
//  - segment_max via order-preserving u32 encoding + global_atomic_max_u32
//    (vectorized: one b128 load + 4 atomics per thread when din % 4 == 0)
//  - per-layer fused GEMM  out = [agg|x] @ [Wl|Wr]^T + b  using
//    v_wmma_f32_16x16x32_f16 (wave32, 16x16 tiles, f32 accumulate)
// ---------------------------------------------------------------------------

typedef __attribute__((ext_vector_type(16))) _Float16 v16h;
typedef __attribute__((ext_vector_type(8)))  _Float16 v8h;
typedef __attribute__((ext_vector_type(8)))  float    v8f;

#define ENC_NEG_INF 0x007FFFFFu   // encoded -inf (ones-complement of 0xFF800000)

static const int N_NODES = 50000;
static const int N_EDGES = 800000;

__device__ __forceinline__ unsigned enc_f32(float f) {
  unsigned u = __float_as_uint(f);
  return (u & 0x80000000u) ? ~u : (u | 0x80000000u);
}
__device__ __forceinline__ float dec_f32(unsigned k) {
  unsigned u = (k & 0x80000000u) ? (k & 0x7FFFFFFFu) : ~k;
  return __uint_as_float(u);
}

// ---------------------------------------------------------------------------
__global__ void fill_u32_k(unsigned* __restrict__ p, long long n, unsigned v) {
  long long i = (long long)blockIdx.x * blockDim.x + threadIdx.x;
  if (i < n) p[i] = v;
}

// vectorized scatter-max: one thread per (edge, 4-feature chunk)
__global__ void scatter_max4_k(const float* __restrict__ x,
                               const int* __restrict__ ei,   // [2, E]
                               unsigned* __restrict__ aggkey,
                               int din, int dinv4, int E, long long total) {
  long long i = (long long)blockIdx.x * blockDim.x + threadIdx.x;
  if (i >= total) return;
  int e = (int)(i / dinv4);
  int c = (int)(i - (long long)e * dinv4);
  int s = ei[e];
  int d = ei[E + e];
  const float4 v = *(const float4*)(x + (long long)s * din + c * 4);
  unsigned* dst = aggkey + (long long)d * din + c * 4;
  atomicMax(dst + 0, enc_f32(v.x));
  atomicMax(dst + 1, enc_f32(v.y));
  atomicMax(dst + 2, enc_f32(v.z));
  atomicMax(dst + 3, enc_f32(v.w));
}

// scalar fallback (din not a multiple of 4, e.g. 131)
__global__ void scatter_max_k(const float* __restrict__ x,
                              const int* __restrict__ ei,
                              unsigned* __restrict__ aggkey,
                              int din, int E, long long total) {
  long long i = (long long)blockIdx.x * blockDim.x + threadIdx.x;
  if (i >= total) return;
  int e = (int)(i / din);
  int f = (int)(i - (long long)e * din);
  int s = ei[e];
  int d = ei[E + e];
  unsigned key = enc_f32(x[(long long)s * din + f]);
  atomicMax(&aggkey[(long long)d * din + f], key);
}

// build f16 AB matrix [N, 2*dinp]: [ agg(-inf->0) | x ], zero-padded K
__global__ void build_ab_k(const unsigned* __restrict__ aggkey,
                           const float* __restrict__ x,
                           _Float16* __restrict__ AB,
                           int din, int dinp, long long total) {
  long long i = (long long)blockIdx.x * blockDim.x + threadIdx.x;
  if (i >= total) return;
  int n = (int)(i / dinp);
  int k = (int)(i - (long long)n * dinp);
  int K = 2 * dinp;
  float av = 0.0f, xv = 0.0f;
  if (k < din) {
    unsigned kk = aggkey[(long long)n * din + k];
    av = (kk == ENC_NEG_INF) ? 0.0f : dec_f32(kk);
    xv = x[(long long)n * din + k];
  }
  AB[(long long)n * K + k]        = (_Float16)av;
  AB[(long long)n * K + dinp + k] = (_Float16)xv;
}

// convert weights to f16 concat layout [doutp, 2*dinp] = [Wl | Wr], pad bias
__global__ void conv_w_k(const float* __restrict__ Wl, const float* __restrict__ bl,
                         const float* __restrict__ Wr,
                         _Float16* __restrict__ Wcat, float* __restrict__ biasp,
                         int din, int dinp, int dout, int doutp, long long total) {
  long long i = (long long)blockIdx.x * blockDim.x + threadIdx.x;
  if (i >= total) return;
  int r = (int)(i / dinp);
  int k = (int)(i - (long long)r * dinp);
  int K = 2 * dinp;
  float wl = 0.0f, wr = 0.0f;
  if (r < dout && k < din) {
    wl = Wl[(long long)r * din + k];
    wr = Wr[(long long)r * din + k];
  }
  Wcat[(long long)r * K + k]        = (_Float16)wl;
  Wcat[(long long)r * K + dinp + k] = (_Float16)wr;
  if (k == 0) biasp[r] = (r < dout) ? bl[r] : 0.0f;
}

// ---------------------------------------------------------------------------
// WMMA GEMM: one wave per 16x16 output tile.  A: AB [N, K] f16 row-major.
// B: Wcat [doutp, K] f16 (column n of B == row n of Wcat -> contiguous loads).
union V16 { v16h v; v8h h[2]; };

__global__ void wmma_gemm_k(const _Float16* __restrict__ AB,
                            const _Float16* __restrict__ Wc,
                            const float* __restrict__ bias,
                            float* __restrict__ out,
                            int K, int dout, int tiles_n, int total_tiles, int relu) {
  int wave = (int)((blockIdx.x * blockDim.x + threadIdx.x) >> 5);
  int lane = threadIdx.x & 31;
  if (wave >= total_tiles) return;          // wave-uniform: EXEC stays all-ones
  int tm = wave / tiles_n;
  int tn = wave - tm * tiles_n;

  int arow  = tm * 16 + (lane & 15);        // A row held by this lane
  int bcol  = tn * 16 + (lane & 15);        // B column held by this lane
  int kbase = (lane < 16) ? 0 : 8;          // A K-sub-offset   (ISA 16-bit A layout)
  int klo   = (lane < 16) ? 0 : 8;          // B K-sub-offset   (same order as A)

  const _Float16* ap = AB + (long long)arow * K + kbase;
  const _Float16* bp = Wc + (long long)bcol * K + klo;

  v8f c = {};
  for (int k = 0; k < K; k += 32) {
    V16 a, b;
    a.h[0] = *(const v8h*)(ap + k);         // K = kb..kb+7
    a.h[1] = *(const v8h*)(ap + k + 16);    // K = 16+kb..16+kb+7
    b.h[0] = *(const v8h*)(bp + k);         // K = klo..klo+7
    b.h[1] = *(const v8h*)(bp + k + 16);    // K = 16+klo..16+klo+7
    c = __builtin_amdgcn_wmma_f32_16x16x32_f16(
            /*neg_a=*/false, a.v, /*neg_b=*/false, b.v,
            /*c_mod=*/(short)0, c, /*reuse_a=*/false, /*reuse_b=*/false);
  }

  float bv = bias[bcol];
  if (bcol < dout) {
    int rbase = tm * 16 + ((lane < 16) ? 0 : 8);
#pragma unroll
    for (int j = 0; j < 8; ++j) {
      float v = c[j] + bv;
      if (relu) v = v > 0.0f ? v : 0.0f;
      out[(long long)(rbase + j) * dout + bcol] = v;
    }
  }
}

// ---------------------------------------------------------------------------
__global__ void logsoftmax3_k(const float* __restrict__ logits,
                              float* __restrict__ out_ls,
                              float* __restrict__ labels, int N) {
  int n = blockIdx.x * blockDim.x + threadIdx.x;
  if (n >= N) return;
  float v0 = logits[n * 3 + 0], v1 = logits[n * 3 + 1], v2 = logits[n * 3 + 2];
  float m = fmaxf(v0, fmaxf(v1, v2));
  float s = expf(v0 - m) + expf(v1 - m) + expf(v2 - m);
  float l = logf(s);
  out_ls[n * 3 + 0] = v0 - m - l;
  out_ls[n * 3 + 1] = v1 - m - l;
  out_ls[n * 3 + 2] = v2 - m - l;
  int idx = 0; float b = v0;                // first-max argmax (jnp semantics)
  if (v1 > b) { b = v1; idx = 1; }
  if (v2 > b) { b = v2; idx = 2; }
  labels[n] = (idx == 2) ? 2.0f : 0.0f;
}

__global__ void concat_k(const float* __restrict__ x, const float* __restrict__ rm,
                         float* __restrict__ xcat, int N) {
  long long i = (long long)blockIdx.x * blockDim.x + threadIdx.x;
  long long total = (long long)N * 131;
  if (i >= total) return;
  int n = (int)(i / 131);
  int k = (int)(i - (long long)n * 131);
  xcat[i] = (k < 128) ? x[(long long)n * 128 + k] : rm[n * 3 + (k - 128)];
}

// final linear [N,64]->[N,2], flatten, multiply by concat(labels,labels) mask
__global__ void joint3_k(const float* __restrict__ h, const float* __restrict__ Wj,
                         const float* __restrict__ bj, const float* __restrict__ labels,
                         float* __restrict__ outj, int N) {
  int i = blockIdx.x * blockDim.x + threadIdx.x;
  if (i >= 2 * N) return;
  int n = i >> 1, c = i & 1;
  float acc = bj[c];
#pragma unroll 8
  for (int k = 0; k < 64; ++k) acc += h[(long long)n * 64 + k] * Wj[c * 64 + k];
  float mask = labels[(i < N) ? i : (i - N)];
  outj[i] = acc * mask;
}

// ---------------------------------------------------------------------------
static void sage_layer(const float* xin, int din,
                       const float* Wl, const float* bl, const float* Wr,
                       int dout, int relu, const int* ei, float* out,
                       unsigned* aggkey, _Float16* AB, _Float16* Wcat, float* biasp,
                       hipStream_t s) {
  const int N = N_NODES, E = N_EDGES;
  int dinp  = (din + 31) & ~31;
  int doutp = (dout + 15) & ~15;
  int K = 2 * dinp;

  long long nk = (long long)N * din;
  fill_u32_k<<<(unsigned)((nk + 255) / 256), 256, 0, s>>>(aggkey, nk, ENC_NEG_INF);

  if ((din & 3) == 0) {
    int dinv4 = din >> 2;
    long long ed = (long long)E * dinv4;
    scatter_max4_k<<<(unsigned)((ed + 255) / 256), 256, 0, s>>>(xin, ei, aggkey,
                                                                din, dinv4, E, ed);
  } else {
    long long ed = (long long)E * din;
    scatter_max_k<<<(unsigned)((ed + 255) / 256), 256, 0, s>>>(xin, ei, aggkey, din, E, ed);
  }

  long long na = (long long)N * dinp;
  build_ab_k<<<(unsigned)((na + 255) / 256), 256, 0, s>>>(aggkey, xin, AB, din, dinp, na);

  long long wk = (long long)doutp * dinp;
  conv_w_k<<<(unsigned)((wk + 255) / 256), 256, 0, s>>>(Wl, bl, Wr, Wcat, biasp,
                                                        din, dinp, dout, doutp, wk);

  int tiles_n = doutp / 16;
  int total_tiles = (N / 16) * tiles_n;     // N = 50000 = 3125 * 16, no M tail
  wmma_gemm_k<<<(total_tiles + 7) / 8, 256, 0, s>>>(AB, Wcat, biasp, out,
                                                    K, dout, tiles_n, total_tiles, relu);
}

extern "C" void kernel_launch(void* const* d_in, const int* in_sizes, int n_in,
                              void* d_out, int out_size, void* d_ws, size_t ws_size,
                              hipStream_t stream) {
  const int N = N_NODES;
  (void)n_in; (void)out_size; (void)ws_size;

  // --- input pointer table (handles insertion-order or jax-sorted pytree) ---
  const float *x, *rm1W, *rm1b, *rm1R, *rm2W, *rm2b, *rm2R, *rm3W, *rm3b, *rm3R,
              *rm4W, *rm4b, *rm4R, *sh1W, *sh1b, *sh1R, *j1W, *j1b, *j1R,
              *j2W, *j2b, *j2R, *j3W, *j3b;
  const int* ei;
  if (in_sizes[0] == 2 * N_EDGES) {
    // sorted keys: edge_index, joint1, joint2, joint3, rm1..rm4, shared1, x
    ei   = (const int*)d_in[0];
    j1W  = (const float*)d_in[1];  j1b = (const float*)d_in[2];  j1R = (const float*)d_in[3];
    j2W  = (const float*)d_in[4];  j2b = (const float*)d_in[5];  j2R = (const float*)d_in[6];
    j3W  = (const float*)d_in[7];  j3b = (const float*)d_in[8];
    rm1W = (const float*)d_in[9];  rm1b = (const float*)d_in[10]; rm1R = (const float*)d_in[11];
    rm2W = (const float*)d_in[12]; rm2b = (const float*)d_in[13]; rm2R = (const float*)d_in[14];
    rm3W = (const float*)d_in[15]; rm3b = (const float*)d_in[16]; rm3R = (const float*)d_in[17];
    rm4W = (const float*)d_in[18]; rm4b = (const float*)d_in[19]; rm4R = (const float*)d_in[20];
    sh1W = (const float*)d_in[21]; sh1b = (const float*)d_in[22]; sh1R = (const float*)d_in[23];
    x    = (const float*)d_in[24];
  } else {
    // insertion order: x, rm1..rm4, shared1, joint1, joint2, joint3, edge_index
    x    = (const float*)d_in[0];
    rm1W = (const float*)d_in[1];  rm1b = (const float*)d_in[2];  rm1R = (const float*)d_in[3];
    rm2W = (const float*)d_in[4];  rm2b = (const float*)d_in[5];  rm2R = (const float*)d_in[6];
    rm3W = (const float*)d_in[7];  rm3b = (const float*)d_in[8];  rm3R = (const float*)d_in[9];
    rm4W = (const float*)d_in[10]; rm4b = (const float*)d_in[11]; rm4R = (const float*)d_in[12];
    sh1W = (const float*)d_in[13]; sh1b = (const float*)d_in[14]; sh1R = (const float*)d_in[15];
    j1W  = (const float*)d_in[16]; j1b = (const float*)d_in[17];  j1R = (const float*)d_in[18];
    j2W  = (const float*)d_in[19]; j2b = (const float*)d_in[20];  j2R = (const float*)d_in[21];
    j3W  = (const float*)d_in[22]; j3b = (const float*)d_in[23];
    ei   = (const int*)d_in[24];
  }

  // --- workspace carve-up ---
  char* ws = (char*)d_ws;
  size_t off = 0;
  auto take = [&](size_t bytes) -> void* {
    void* p = ws + off;
    off += (bytes + 255) & ~(size_t)255;
    return p;
  };
  unsigned*  aggkey = (unsigned*)take((size_t)N * 160 * 4);
  _Float16*  AB     = (_Float16*)take((size_t)N * 320 * 2);
  _Float16*  Wcat   = (_Float16*)take((size_t)128 * 320 * 2);
  float*     biasp  = (float*)take(128 * 4);
  float*     B0     = (float*)take((size_t)N * 131 * 4);
  float*     B1     = (float*)take((size_t)N * 131 * 4);
  float*     B2     = (float*)take((size_t)N * 131 * 4);
  float*     rmlog  = (float*)take((size_t)N * 3 * 4);
  float*     labels = (float*)take((size_t)N * 4);

  float* out_ls = (float*)d_out;                   // [N,3] log-softmax
  float* out_j  = (float*)d_out + (size_t)N * 3;   // [2N] masked joint output

  // --- rm branch ---
  sage_layer(x,  128, rm1W, rm1b, rm1R, 128, 1, ei, B0, aggkey, AB, Wcat, biasp, stream);
  sage_layer(B0, 128, rm2W, rm2b, rm2R, 128, 1, ei, B1, aggkey, AB, Wcat, biasp, stream);
  sage_layer(B1, 128, rm3W, rm3b, rm3R,  64, 1, ei, B0, aggkey, AB, Wcat, biasp, stream);
  sage_layer(B0,  64, rm4W, rm4b, rm4R,   3, 0, ei, rmlog, aggkey, AB, Wcat, biasp, stream);

  logsoftmax3_k<<<(N + 255) / 256, 256, 0, stream>>>(rmlog, out_ls, labels, N);

  // --- shared layer on concat(x, rm) ---
  long long ct = (long long)N * 131;
  concat_k<<<(unsigned)((ct + 255) / 256), 256, 0, stream>>>(x, rmlog, B2, N);
  sage_layer(B2, 131, sh1W, sh1b, sh1R, 128, 1, ei, B0, aggkey, AB, Wcat, biasp, stream);

  // --- joint branch ---
  sage_layer(B0, 128, j1W, j1b, j1R, 128, 1, ei, B1, aggkey, AB, Wcat, biasp, stream);
  sage_layer(B1, 128, j2W, j2b, j2R,  64, 1, ei, B2, aggkey, AB, Wcat, biasp, stream);

  joint3_k<<<(2 * N + 255) / 256, 256, 0, stream>>>(B2, j3W, j3b, labels, out_j, N);
}
